// PseudoLinearModel_41618233098550
// MI455X (gfx1250) — matrix operand order repaired and checked by reading, not compiled, not run on hardware
//
#include <hip/hip_runtime.h>

// ---------------------------------------------------------------------------
// CDNA5 (gfx1250) fused periodic-CNN stencil model.
//   cvt_* kernels : f32 -> f16 weight transposes into d_ws
//   conv0_k       : 1->32 scalar conv + relu -> hA (f16 NHWC)
//   conv32_wmma x3: 32->32 implicit-GEMM WMMA convs (relu), LDS-staged weights
//   final_k       : conv4 (32->80 padded) WMMA + constraint einsum WMMA +
//                   stencil dot -> d_out f32
// All hot-loop addressing is 32-bit (buffers < 2 GB) so global loads use the
// SGPR-base + VGPR-offset form instead of 64-bit VGPR address math.
// ---------------------------------------------------------------------------

typedef __attribute__((ext_vector_type(16))) _Float16 v16h;
typedef __attribute__((ext_vector_type(8)))  _Float16 v8h;
typedef __attribute__((ext_vector_type(8)))  float    v8f;
typedef __attribute__((ext_vector_type(4)))  int      v4i;

#define HW   256
#define NB   16
#define CH   32
#define TAPS 25

// ---- async global->LDS staging (gfx1250 ASYNCcnt path), with safe fallback --
#if __has_builtin(__builtin_amdgcn_global_load_async_to_lds_b128)
#define HAVE_ASYNC_LDS 1
#endif

static __device__ __forceinline__ void stage16(const void* g, void* l) {
#if defined(HAVE_ASYNC_LDS)
  __builtin_amdgcn_global_load_async_to_lds_b128((v4i*)g, (v4i*)l, 0, 0);
#else
  *(v8h*)l = *(const v8h*)g;
#endif
}

static __device__ __forceinline__ void stage_wait() {
#if defined(HAVE_ASYNC_LDS)
#if __has_builtin(__builtin_amdgcn_s_wait_asynccnt)
  __builtin_amdgcn_s_wait_asynccnt(0);
#else
  asm volatile("s_wait_asynccnt 0x0" ::: "memory");
#endif
#endif
}

union ABfrag { v16h v; v8h h[2]; };

// A-fragment (16x32 f16): lane half h holds K h*8..h*8+7 and 16+h*8..16+h*8+7
static __device__ __forceinline__ v16h load_afrag(const _Float16* base16B) {
  ABfrag u;
  u.h[0] = *(const v8h*)(base16B);
  u.h[1] = *(const v8h*)(base16B + 16);
  return u.v;
}
// B-fragment (32x16 f16): lane half h holds K h*8..h*8+7 and 16+h*8..16+h*8+7 for column N
static __device__ __forceinline__ v16h load_bfrag(const _Float16* base16B) {
  ABfrag u;
  u.h[0] = *(const v8h*)(base16B);
  u.h[1] = *(const v8h*)(base16B + 16);
  return u.v;
}

// ----------------------- weight conversion kernels -------------------------

// w[K,K,32,32] (k1,k2,ic,oc) -> wt[tap][oc][ic] f16
__global__ void cvt_w123(const float* __restrict__ w, _Float16* __restrict__ wt) {
  int i = blockIdx.x * 256 + threadIdx.x;
  if (i >= TAPS * 32 * 32) return;
  int ic = i & 31, oc = (i >> 5) & 31, t = i >> 10;
  wt[i] = (_Float16)w[(t * 32 + ic) * 32 + oc];
}

// w4[K,K,32,75] -> w4t[tap][oc(80 padded)][ic] f16
__global__ void cvt_w4(const float* __restrict__ w, _Float16* __restrict__ wt) {
  int i = blockIdx.x * 256 + threadIdx.x;
  if (i >= TAPS * 80 * 32) return;
  int ic = i & 31, oc = (i >> 5) % 80, t = i / (80 * 32);
  float v = (oc < 75) ? w[(t * 32 + ic) * 75 + oc] : 0.f;
  wt[i] = (_Float16)v;
}

// P[3,25,25] (d,k,s) -> Pt[d][s(32 padded)][k(32 padded)] f16
__global__ void cvt_P(const float* __restrict__ P, _Float16* __restrict__ Pt) {
  int i = blockIdx.x * 256 + threadIdx.x;
  if (i >= 3 * 32 * 32) return;
  int k = i & 31, s = (i >> 5) & 31, d = i >> 10;
  float v = (k < 25 && s < 25) ? P[(d * 25 + k) * 25 + s] : 0.f;
  Pt[i] = (_Float16)v;
}

// ----------------------------- conv0 (1->32) --------------------------------

__global__ __launch_bounds__(256)
void conv0_k(const float* __restrict__ x, const float* __restrict__ w0,
             const float* __restrict__ b0, _Float16* __restrict__ hout) {
  __shared__ float w0s[TAPS * 32];               // 3200 B = 200 x 16 B
  if (threadIdx.x < 200)
    stage16(w0 + threadIdx.x * 4, &w0s[threadIdx.x * 4]);
  stage_wait();
  __syncthreads();

  int p = blockIdx.x * 256 + threadIdx.x;         // pixel id over B*H*W (< 2^20)
  int b = p >> 16, y = (p >> 8) & 255, xw = p & 255;
  int ibase = (b * HW) * HW;
  float patch[TAPS];
#pragma unroll
  for (int t = 0; t < TAPS; ++t) {
    int ys = (y + t / 5 - 2) & 255;
    int xs = (xw + t % 5 - 2) & 255;
    patch[t] = x[ibase + ys * HW + xs];
  }
  float a[32];
#pragma unroll
  for (int oc = 0; oc < 32; ++oc) {
    float s = b0[oc];
#pragma unroll
    for (int t = 0; t < TAPS; ++t) s += patch[t] * w0s[t * 32 + oc];
    a[oc] = s > 0.f ? s : 0.f;
  }
#pragma unroll
  for (int j = 0; j < 4; ++j) {
    v8h pack;
#pragma unroll
    for (int e = 0; e < 8; ++e) pack[e] = (_Float16)a[j * 8 + e];
    *(v8h*)(hout + p * 32 + j * 8) = pack;
  }
}

// ------------------------ conv 32->32 implicit GEMM -------------------------

__global__ __launch_bounds__(256)
void conv32_wmma(const _Float16* __restrict__ hin, const _Float16* __restrict__ wt,
                 const float* __restrict__ bias, _Float16* __restrict__ hout) {
  __shared__ _Float16 wlds[TAPS * 32 * 32];  // [tap][oc][ic], 50 KB
  __shared__ _Float16 olds[8][32 * 32];      // per-wave output staging, 16 KB
  int tid = threadIdx.x;
  for (int i = tid; i < TAPS * 32 * 32 / 8; i += 256)
    stage16(wt + i * 8, &wlds[i * 8]);
  stage_wait();
  __syncthreads();

  int wave = tid >> 5, lane = tid & 31;
  int lhalf = lane >> 4, l16 = lane & 15;
  int row = blockIdx.x;                // b*256 + y
  int b = row >> 8, y = row & 255;
  int x0 = wave * 32;
  int ibase = (b * HW) * HW;           // element index base, fits in int

  v8f acc[2][2] = {};

#pragma unroll
  for (int t = 0; t < TAPS; ++t) {
    int dy = t / 5 - 2, dx = t % 5 - 2;
    int rowbase = ibase + ((y + dy) & 255) * HW;

    int xs0 = (x0 + l16 + dx) & 255;
    int xs1 = (x0 + 16 + l16 + dx) & 255;
    v16h a0 = load_afrag(hin + (rowbase + xs0) * CH + lhalf * 8);
    v16h a1 = load_afrag(hin + (rowbase + xs1) * CH + lhalf * 8);
    v16h b0f = load_bfrag(&wlds[(t * 32 + l16) * 32 + lhalf * 8]);
    v16h b1f = load_bfrag(&wlds[(t * 32 + 16 + l16) * 32 + lhalf * 8]);

    acc[0][0] = __builtin_amdgcn_wmma_f32_16x16x32_f16(false, a0, false, b0f, (short)0, acc[0][0], false, false);
    acc[0][1] = __builtin_amdgcn_wmma_f32_16x16x32_f16(false, a0, false, b1f, (short)0, acc[0][1], false, false);
    acc[1][0] = __builtin_amdgcn_wmma_f32_16x16x32_f16(false, a1, false, b0f, (short)0, acc[1][0], false, false);
    acc[1][1] = __builtin_amdgcn_wmma_f32_16x16x32_f16(false, a1, false, b1f, (short)0, acc[1][1], false, false);
  }

  // epilogue: bias + relu -> per-wave LDS tile -> contiguous b128 stores
#pragma unroll
  for (int mt = 0; mt < 2; ++mt) {
#pragma unroll
    for (int nt = 0; nt < 2; ++nt) {
      int oc = nt * 16 + l16;
      float bv = bias[oc];
#pragma unroll
      for (int v = 0; v < 8; ++v) {
        int m = mt * 16 + lhalf * 8 + v;     // D layout: lanes>=16 hold M=v+8
        float r = acc[mt][nt][v] + bv;
        r = r > 0.f ? r : 0.f;
        olds[wave][m * 32 + oc] = (_Float16)r;
      }
    }
  }
  int base = (ibase + y * HW + x0) * CH;      // contiguous 2 KB span
#pragma unroll
  for (int i = lane; i < 32 * 32 / 8; i += 32)
    *(v8h*)(hout + base + i * 8) = *(const v8h*)&olds[wave][i * 8];
}

// ------------- conv4 (32->80) + constraint einsum + stencil dot -------------

__global__ __launch_bounds__(256)
void final_k(const _Float16* __restrict__ hin, const _Float16* __restrict__ w4t,
             const float* __restrict__ b4, const _Float16* __restrict__ Pt,
             const float* __restrict__ cb, const float* __restrict__ x,
             float* __restrict__ out) {
  __shared__ _Float16 netlds[8][32 * 96];  // per-wave: 32 px x (3 d x 32 padded k)
  __shared__ _Float16 plds[3 * 32 * 32];   // padded P, 6 KB
  int tid = threadIdx.x;
  for (int i = tid; i < 3 * 32 * 32 / 8; i += 256)
    stage16(Pt + i * 8, &plds[i * 8]);
  for (int i = tid; i < 8 * 32 * 96; i += 256) ((_Float16*)netlds)[i] = (_Float16)0.f;
  stage_wait();
  __syncthreads();

  int wave = tid >> 5, lane = tid & 31;
  int lhalf = lane >> 4, l16 = lane & 15;
  int row = blockIdx.x;
  int b = row >> 8, y = row & 255;
  int x0 = wave * 32;
  int ibase = (b * HW) * HW;

  // ---- conv4: 32 px x 80 oc, K = 25 taps x 32 ch ----
  v8f acc[2][5] = {};
#pragma unroll 5
  for (int t = 0; t < TAPS; ++t) {
    int dy = t / 5 - 2, dx = t % 5 - 2;
    int rowbase = ibase + ((y + dy) & 255) * HW;

    int xs0 = (x0 + l16 + dx) & 255;
    int xs1 = (x0 + 16 + l16 + dx) & 255;
    v16h a0 = load_afrag(hin + (rowbase + xs0) * CH + lhalf * 8);
    v16h a1 = load_afrag(hin + (rowbase + xs1) * CH + lhalf * 8);
#pragma unroll
    for (int nt = 0; nt < 5; ++nt) {
      v16h bf = load_bfrag(w4t + (t * 80 + nt * 16 + l16) * 32 + lhalf * 8);
      acc[0][nt] = __builtin_amdgcn_wmma_f32_16x16x32_f16(false, a0, false, bf, (short)0, acc[0][nt], false, false);
      acc[1][nt] = __builtin_amdgcn_wmma_f32_16x16x32_f16(false, a1, false, bf, (short)0, acc[1][nt], false, false);
    }
  }

  // ---- scatter net (+bias) into LDS as padded [px][d*32+kk] fp16 ----
#pragma unroll
  for (int mt = 0; mt < 2; ++mt) {
#pragma unroll
    for (int nt = 0; nt < 5; ++nt) {
      int oc = nt * 16 + l16;
      if (oc < 75) {
        float bv = b4[oc];
        int d = oc / 25, kk = oc % 25;
#pragma unroll
        for (int v = 0; v < 8; ++v) {
          int m = mt * 16 + lhalf * 8 + v;
          netlds[wave][m * 96 + d * 32 + kk] = (_Float16)(acc[mt][nt][v] + bv);
        }
      }
    }
  }
  __syncthreads();

  int rowout = ibase + y * HW;

  // ---- constraint einsum + cb + stencil dot, per derivative d ----
  for (int d = 0; d < 3; ++d) {
    v8f cacc[2][2] = {};
#pragma unroll
    for (int mt = 0; mt < 2; ++mt) {
      v16h af = load_afrag(&netlds[wave][(mt * 16 + l16) * 96 + d * 32 + lhalf * 8]);
#pragma unroll
      for (int nt = 0; nt < 2; ++nt) {
        v16h bf = load_bfrag(&plds[(d * 32 + nt * 16 + l16) * 32 + lhalf * 8]);
        cacc[mt][nt] = __builtin_amdgcn_wmma_f32_16x16x32_f16(false, af, false, bf, (short)0, cacc[mt][nt], false, false);
      }
    }

    // reduce over stencil index s = nt*16 + l16 (zero cols for s>=25)
    float red[2][8] = {};
#pragma unroll
    for (int mt = 0; mt < 2; ++mt) {
#pragma unroll
      for (int nt = 0; nt < 2; ++nt) {
        int s = nt * 16 + l16;
        float cbv = (s < 25) ? cb[d * 25 + s] : 0.f;
        int dH = s / 5 - 2, dW = s % 5 - 2;
        int pbase = ibase + ((y + dH) & 255) * HW;
#pragma unroll
        for (int v = 0; v < 8; ++v) {
          int m = mt * 16 + lhalf * 8 + v;
          float pv = 0.f;
          if (s < 25) pv = x[pbase + ((x0 + m + dW) & 255)];
          red[mt][v] += (cacc[mt][nt][v] + cbv) * pv;
        }
      }
    }
    // butterfly within each 16-lane half (halves hold different rows M)
#pragma unroll
    for (int mask = 1; mask <= 8; mask <<= 1)
#pragma unroll
      for (int mt = 0; mt < 2; ++mt)
#pragma unroll
        for (int v = 0; v < 8; ++v)
          red[mt][v] += __shfl_xor(red[mt][v], mask, 32);

    if (l16 == 0) {  // lanes 0 (rows M=v) and 16 (rows M=v+8) write
#pragma unroll
      for (int mt = 0; mt < 2; ++mt)
#pragma unroll
        for (int v = 0; v < 8; ++v) {
          int m = mt * 16 + lhalf * 8 + v;
          __builtin_nontemporal_store(red[mt][v], out + (rowout + x0 + m) * 3 + d);
        }
    }
  }
}

// ------------------------------- launcher -----------------------------------

extern "C" void kernel_launch(void* const* d_in, const int* in_sizes, int n_in,
                              void* d_out, int out_size, void* d_ws, size_t ws_size,
                              hipStream_t stream) {
  const float* x  = (const float*)d_in[0];
  const float* w0 = (const float*)d_in[1];
  const float* b0 = (const float*)d_in[2];
  const float* w1 = (const float*)d_in[3];
  const float* b1 = (const float*)d_in[4];
  const float* w2 = (const float*)d_in[5];
  const float* b2 = (const float*)d_in[6];
  const float* w3 = (const float*)d_in[7];
  const float* b3 = (const float*)d_in[8];
  const float* w4 = (const float*)d_in[9];
  const float* b4 = (const float*)d_in[10];
  const float* P  = (const float*)d_in[11];
  const float* cb = (const float*)d_in[12];

  char* ws = (char*)d_ws;
  const size_t actBytes = (size_t)NB * HW * HW * CH * sizeof(_Float16);  // 64 MB
  _Float16* hA  = (_Float16*)(ws);
  _Float16* hB  = (_Float16*)(ws + actBytes);
  _Float16* wt1 = (_Float16*)(ws + 2 * actBytes);
  _Float16* wt2 = wt1 + TAPS * 32 * 32;
  _Float16* wt3 = wt2 + TAPS * 32 * 32;
  _Float16* w4t = wt3 + TAPS * 32 * 32;
  _Float16* Ptr = w4t + TAPS * 80 * 32;

  cvt_w123<<<(TAPS * 32 * 32 + 255) / 256, 256, 0, stream>>>(w1, wt1);
  cvt_w123<<<(TAPS * 32 * 32 + 255) / 256, 256, 0, stream>>>(w2, wt2);
  cvt_w123<<<(TAPS * 32 * 32 + 255) / 256, 256, 0, stream>>>(w3, wt3);
  cvt_w4  <<<(TAPS * 80 * 32 + 255) / 256, 256, 0, stream>>>(w4, w4t);
  cvt_P   <<<(3 * 32 * 32   + 255) / 256, 256, 0, stream>>>(P,  Ptr);

  const int rows = NB * HW;  // 4096 blocks, one image row each
  conv0_k    <<<rows, 256, 0, stream>>>(x, w0, b0, hA);
  conv32_wmma<<<rows, 256, 0, stream>>>(hA, wt1, b1, hB);
  conv32_wmma<<<rows, 256, 0, stream>>>(hB, wt2, b2, hA);
  conv32_wmma<<<rows, 256, 0, stream>>>(hA, wt3, b3, hB);
  final_k    <<<rows, 256, 0, stream>>>(hB, w4t, b4, Ptr, cb, x, (float*)d_out);
}
